// MultiHeadAttention_15891378995335
// MI455X (gfx1250) — matrix operand order repaired and checked by reading, not compiled, not run on hardware
//
#include <hip/hip_runtime.h>

// ---------------------------------------------------------------------------
// MHA forward for gfx1250 (MI455X), bf16 WMMA pipeline.
// B=2, S=2048, D=1024, H=16, dh=64 (hard-coded from the reference).
// GEMMs use 32x64 register blocking per wave (8 WMMAs / k-step).
// ---------------------------------------------------------------------------

typedef __bf16 bf16;
typedef __attribute__((ext_vector_type(16))) bf16  v16bf;
typedef __attribute__((ext_vector_type(8)))  bf16  v8bf;
typedef __attribute__((ext_vector_type(8)))  float v8f;

#define Bdim 2
#define Sdim 2048
#define Ddim 1024
#define Hdim 16
#define DH   64

static __device__ inline v8f wmma_bf16(v16bf a, v16bf b, v8f c) {
  return __builtin_amdgcn_wmma_f32_16x16x32_bf16(
      /*neg_a=*/false, a, /*neg_b=*/false, b,
      /*c_mod=*/(short)0, c, /*reuse_a=*/false, /*reuse_b=*/false);
}

// Assemble A-operand (contiguous k order, same as B; one 32-K chunk):
// lane<16: K = {kb..kb+15}, lane>=16: +16. Caller passes
// row pointer already offset by (chunk_base + (lane>>4)*16).
static __device__ inline v16bf load_a16(const bf16* p) {
  union { v16bf v; v8bf h[2]; } u;
  u.h[0] = *(const v8bf*)(p);
  u.h[1] = *(const v8bf*)(p + 8);
  return u.v;
}

static __device__ inline float half_max16(float x) {
  x = fmaxf(x, __shfl_xor(x, 1, 32));
  x = fmaxf(x, __shfl_xor(x, 2, 32));
  x = fmaxf(x, __shfl_xor(x, 4, 32));
  x = fmaxf(x, __shfl_xor(x, 8, 32));
  return x;
}
static __device__ inline float half_sum16(float x) {
  x += __shfl_xor(x, 1, 32);
  x += __shfl_xor(x, 2, 32);
  x += __shfl_xor(x, 4, 32);
  x += __shfl_xor(x, 8, 32);
  return x;
}

// ---------------------------------------------------------------------------
// 1) elementwise f32 -> bf16
// ---------------------------------------------------------------------------
__global__ __launch_bounds__(256) void cast_f32_bf16(const float* __restrict__ in,
                                                     bf16* __restrict__ out, int n) {
  int i = blockIdx.x * 256 + threadIdx.x;
  if (i < n) out[i] = (bf16)in[i];
}

// 2) transpose + convert: out[n*K + k] = w[k*N + n]  (w is K x N row-major)
__global__ __launch_bounds__(256) void transpose_f32_bf16(const float* __restrict__ w,
                                                          bf16* __restrict__ out,
                                                          int K, int N) {
  int idx = blockIdx.x * 256 + threadIdx.x;
  if (idx >= K * N) return;
  int n = idx / K;
  int k = idx % K;
  out[(size_t)n * K + k] = (bf16)w[(size_t)k * N + n];
}

// ---------------------------------------------------------------------------
// 3) QKV GEMM: [4096 x 3072] = Xbf16[4096 x 1024] @ Wattn (B from N-major Wt).
//    32x64 output tile per wave; scatters q/k/v into [B,H,S,dh] bf16.
//    N-tile is 64-wide => q/k/v selector and head index are wave-uniform.
// ---------------------------------------------------------------------------
__global__ __launch_bounds__(256) void qkv_gemm(const bf16* __restrict__ X,
                                                const bf16* __restrict__ Wt,
                                                const float* __restrict__ bias,
                                                bf16* __restrict__ Q,
                                                bf16* __restrict__ K,
                                                bf16* __restrict__ V) {
  const int wave = threadIdx.x >> 5;
  const int lane = threadIdx.x & 31;
  const int tile = blockIdx.x * 8 + wave;        // 6144 tiles total
  const int tm = tile / 48;                      // 4096/32 = 128 row tiles
  const int tn = tile % 48;                      // 3072/64 = 48  col tiles
  const int am = lane & 15;
  const int ah = lane >> 4;

  const bf16* Arow0 = X + (size_t)(tm * 32 + am) * Ddim;
  const bf16* Arow1 = X + (size_t)(tm * 32 + 16 + am) * Ddim;
  const bf16* Bc[4];
#pragma unroll
  for (int t = 0; t < 4; ++t)
    Bc[t] = Wt + (size_t)(tn * 64 + t * 16 + am) * Ddim;

  v8f acc[2][4];
#pragma unroll
  for (int mt = 0; mt < 2; ++mt)
#pragma unroll
    for (int t = 0; t < 4; ++t) acc[mt][t] = (v8f){};

#pragma unroll 2
  for (int k0 = 0; k0 < Ddim; k0 += 32) {
    v16bf a0 = load_a16(Arow0 + k0 + ah * 16);
    v16bf a1 = load_a16(Arow1 + k0 + ah * 16);
#pragma unroll
    for (int t = 0; t < 4; ++t) {
      v16bf b = *(const v16bf*)(Bc[t] + k0 + ah * 16);
      acc[0][t] = wmma_bf16(a0, b, acc[0][t]);
      acc[1][t] = wmma_bf16(a1, b, acc[1][t]);
    }
  }

  // wave-uniform: which buffer + which head (64-col tile never crosses them)
  const int which = tn >> 4;                     // 0=q 1=k 2=v
  const int hh    = tn & 15;                     // head
  bf16* dst = (which == 0) ? Q : (which == 1) ? K : V;
  float bv[4];
#pragma unroll
  for (int t = 0; t < 4; ++t) bv[t] = bias[tn * 64 + t * 16 + am];

#pragma unroll
  for (int mt = 0; mt < 2; ++mt) {
#pragma unroll
    for (int j = 0; j < 8; ++j) {
      int m  = tm * 32 + mt * 16 + j + ah * 8;   // global row = b*S + s
      int bi = m >> 11;
      int si = m & 2047;
      size_t base = (((size_t)bi * Hdim + hh) * Sdim + si) * DH;
#pragma unroll
      for (int t = 0; t < 4; ++t)
        dst[base + t * 16 + am] = (bf16)(acc[mt][t][j] + bv[t]);
    }
  }
}

// ---------------------------------------------------------------------------
// 4) Streaming (flash-style) attention. One wave = 16 queries x dh=64,
//    4 waves per block, per-wave LDS tiles for P-transpose and V^T.
// ---------------------------------------------------------------------------
__global__ __launch_bounds__(128) void attention(const bf16* __restrict__ Qg,
                                                 const bf16* __restrict__ Kg,
                                                 const bf16* __restrict__ Vg,
                                                 bf16* __restrict__ attn) {
  __shared__ __align__(32) bf16 ldsP[4 * 16 * 32];   // per-wave 16x32 P tile
  __shared__ __align__(32) bf16 ldsVT[4 * 64 * 32];  // per-wave V^T (d x key)

  const int wave = threadIdx.x >> 5;
  const int lane = threadIdx.x & 31;
  const int task = blockIdx.x * 4 + wave;   // 0 .. 4095
  const int qt = task & 127;                // 128 query tiles of 16
  const int bh = task >> 7;                 // 0 .. 31
  const int h  = bh & 15;
  const int b  = bh >> 4;
  const int am = lane & 15;
  const int ah = lane >> 4;

  const bf16* Q = Qg + ((size_t)bh * Sdim + qt * 16) * DH;
  const bf16* K = Kg + (size_t)bh * Sdim * DH;
  const bf16* V = Vg + (size_t)bh * Sdim * DH;

  bf16* Pw = ldsP  + wave * (16 * 32);
  bf16* Vt = ldsVT + wave * (64 * 32);

  // Preload the 16x64 Q tile in A layout (two 32-K chunks).
  const bf16* Qrow = Q + am * DH;
  v16bf aq0 = load_a16(Qrow + 0  + ah * 16);
  v16bf aq1 = load_a16(Qrow + 32 + ah * 16);

  float mrow[8], lrow[8];
  v8f accv[4];
#pragma unroll
  for (int j = 0; j < 8; ++j) { mrow[j] = -1e30f; lrow[j] = 0.f; }
#pragma unroll
  for (int t = 0; t < 4; ++t) accv[t] = (v8f){};

  const float scale = 0.125f;  // 1/sqrt(64)

  for (int kb0 = 0; kb0 < Sdim; kb0 += 32) {
    // ---- scores: two 16-key N-tiles, K rows are already B-layout ----
    v8f s0 = {}, s1 = {};
    {
      const bf16* K0 = K + (size_t)(kb0 + am) * DH;
      const bf16* K1 = K + (size_t)(kb0 + 16 + am) * DH;
      v16bf b00 = *(const v16bf*)(K0 + ah * 16);       // d 0..31
      v16bf b01 = *(const v16bf*)(K0 + 32 + ah * 16);  // d 32..63
      v16bf b10 = *(const v16bf*)(K1 + ah * 16);
      v16bf b11 = *(const v16bf*)(K1 + 32 + ah * 16);
      s0 = wmma_bf16(aq0, b00, s0);
      s0 = wmma_bf16(aq1, b01, s0);
      s1 = wmma_bf16(aq0, b10, s1);
      s1 = wmma_bf16(aq1, b11, s1);
    }

    // ---- online softmax update; write P (bf16) into LDS in row-major ----
#pragma unroll
    for (int j = 0; j < 8; ++j) {
      float v0 = s0[j] * scale;
      float v1 = s1[j] * scale;
      float mx = half_max16(fmaxf(v0, v1));
      float mnew = fmaxf(mrow[j], mx);
      float corr = __expf(mrow[j] - mnew);
      float p0 = __expf(v0 - mnew);
      float p1 = __expf(v1 - mnew);
      float rs = half_sum16(p0 + p1);
      lrow[j] = lrow[j] * corr + rs;
      mrow[j] = mnew;
#pragma unroll
      for (int t = 0; t < 4; ++t) accv[t][j] *= corr;
      int prow = j + ah * 8;                 // C-layout row for this lane-half
      Pw[prow * 32 + am]      = (bf16)p0;
      Pw[prow * 32 + 16 + am] = (bf16)p1;
    }

    // ---- stage V block transposed: Vt[d][kj] = V[kb0+kj][d] ----
    {
      const bf16* Vrow = V + (size_t)(kb0 + lane) * DH;
#pragma unroll
      for (int c8 = 0; c8 < 8; ++c8) {
        v8bf vv = *(const v8bf*)(Vrow + c8 * 8);
#pragma unroll
        for (int e = 0; e < 8; ++e) Vt[(c8 * 8 + e) * 32 + lane] = vv[e];
      }
    }
    __syncthreads();

    // ---- PV: A = P (16x32) from LDS, B = Vt columns (contiguous) ----
    {
      v16bf pa = load_a16(Pw + am * 32 + ah * 16);
#pragma unroll
      for (int t = 0; t < 4; ++t) {
        v16bf bv = *(const v16bf*)(Vt + (t * 16 + am) * 32 + ah * 16);
        accv[t] = wmma_bf16(pa, bv, accv[t]);
      }
    }
    __syncthreads();
  }

  // ---- normalize and write merged-head output [B,S,D] bf16 ----
#pragma unroll
  for (int t = 0; t < 4; ++t) {
#pragma unroll
    for (int j = 0; j < 8; ++j) {
      float o = accv[t][j] / lrow[j];
      int m = j + ah * 8;
      int d = t * 16 + am;
      int sg = qt * 16 + m;
      attn[((size_t)b * Sdim + sg) * Ddim + h * DH + d] = (bf16)o;
    }
  }
}

// ---------------------------------------------------------------------------
// 5) output projection: out_f32[4096 x 1024] = attn_bf16 @ Wproj + b_proj.
//    32x64 output tile per wave.
// ---------------------------------------------------------------------------
__global__ __launch_bounds__(256) void proj_gemm(const bf16* __restrict__ A,
                                                 const bf16* __restrict__ Wt,
                                                 const float* __restrict__ bias,
                                                 float* __restrict__ out) {
  const int wave = threadIdx.x >> 5;
  const int lane = threadIdx.x & 31;
  const int tile = blockIdx.x * 8 + wave;   // 2048 tiles
  const int tm = tile / 16;                 // 4096/32 = 128 row tiles
  const int tn = tile % 16;                 // 1024/64 = 16  col tiles
  const int am = lane & 15;
  const int ah = lane >> 4;

  const bf16* Arow0 = A + (size_t)(tm * 32 + am) * Ddim;
  const bf16* Arow1 = A + (size_t)(tm * 32 + 16 + am) * Ddim;
  const bf16* Bc[4];
#pragma unroll
  for (int t = 0; t < 4; ++t)
    Bc[t] = Wt + (size_t)(tn * 64 + t * 16 + am) * Ddim;

  v8f acc[2][4];
#pragma unroll
  for (int mt = 0; mt < 2; ++mt)
#pragma unroll
    for (int t = 0; t < 4; ++t) acc[mt][t] = (v8f){};

#pragma unroll 2
  for (int k0 = 0; k0 < Ddim; k0 += 32) {
    v16bf a0 = load_a16(Arow0 + k0 + ah * 16);
    v16bf a1 = load_a16(Arow1 + k0 + ah * 16);
#pragma unroll
    for (int t = 0; t < 4; ++t) {
      v16bf b = *(const v16bf*)(Bc[t] + k0 + ah * 16);
      acc[0][t] = wmma_bf16(a0, b, acc[0][t]);
      acc[1][t] = wmma_bf16(a1, b, acc[1][t]);
    }
  }

  float bv[4];
#pragma unroll
  for (int t = 0; t < 4; ++t) bv[t] = bias[tn * 64 + t * 16 + am];

#pragma unroll
  for (int mt = 0; mt < 2; ++mt) {
#pragma unroll
    for (int j = 0; j < 8; ++j) {
      int m = tm * 32 + mt * 16 + j + ah * 8;
#pragma unroll
      for (int t = 0; t < 4; ++t)
        out[(size_t)m * Ddim + tn * 64 + t * 16 + am] = acc[mt][t][j] + bv[t];
    }
  }
}

// ---------------------------------------------------------------------------
extern "C" void kernel_launch(void* const* d_in, const int* in_sizes, int n_in,
                              void* d_out, int out_size, void* d_ws, size_t ws_size,
                              hipStream_t stream) {
  (void)in_sizes; (void)n_in; (void)out_size; (void)ws_size;

  const float* x      = (const float*)d_in[0];  // [2,2048,1024]
  const float* w_attn = (const float*)d_in[1];  // [1024,3072]
  const float* b_attn = (const float*)d_in[2];  // [3072]
  const float* w_proj = (const float*)d_in[3];  // [1024,1024]
  const float* b_proj = (const float*)d_in[4];  // [1024]
  float* out = (float*)d_out;

  const size_t nX   = (size_t)Bdim * Sdim * Ddim;   // 4,194,304
  const size_t nWA  = (size_t)Ddim * 3 * Ddim;      // 3,145,728
  const size_t nWP  = (size_t)Ddim * Ddim;          // 1,048,576

  char* ws = (char*)d_ws;
  bf16* xb    = (bf16*)ws;                 ws += nX  * sizeof(bf16);
  bf16* wAt   = (bf16*)ws;                 ws += nWA * sizeof(bf16);
  bf16* wPt   = (bf16*)ws;                 ws += nWP * sizeof(bf16);
  bf16* qb    = (bf16*)ws;                 ws += nX  * sizeof(bf16);
  bf16* kb    = (bf16*)ws;                 ws += nX  * sizeof(bf16);
  bf16* vb    = (bf16*)ws;                 ws += nX  * sizeof(bf16);
  bf16* attnb = (bf16*)ws;                 ws += nX  * sizeof(bf16);

  cast_f32_bf16<<<(int)(nX / 256), 256, 0, stream>>>(x, xb, (int)nX);
  transpose_f32_bf16<<<(int)(nWA / 256), 256, 0, stream>>>(w_attn, wAt, Ddim, 3 * Ddim);
  transpose_f32_bf16<<<(int)(nWP / 256), 256, 0, stream>>>(w_proj, wPt, Ddim, Ddim);

  // 128 x 48 tiles (32x64 each), 8 waves/block
  qkv_gemm<<<768, 256, 0, stream>>>(xb, wAt, b_attn, qb, kb, vb);

  // B*H*(S/16) = 4096 wave-tasks, 4 waves/block
  attention<<<1024, 128, 0, stream>>>(qb, kb, vb, attnb);

  // 128 x 16 tiles (32x64 each), 8 waves/block
  proj_gemm<<<256, 256, 0, stream>>>(attnb, wPt, b_proj, out);
}
